// MultiHeadAttention_6210522710236
// MI455X (gfx1250) — matrix operand order repaired and checked by reading, not compiled, not run on hardware
//
#include <hip/hip_runtime.h>
#include <hip/hip_bf16.h>

// MHA for MI455X (gfx1250): bf16 WMMA + fused LDS-resident attention.
// Stage 1: Q/K projections -> bf16 [B*H][S][Dk]; V projection -> bf16 [B*H][Dk][S] (transposed)
// Stage 2 (fused, one 256-thread block per 16-row query strip):
//          scores strip (16x2048 f32) -> LDS, float4 softmax in LDS,
//          attn written to HBM exactly once (float4 stores),
//          O strip = P @ V from LDS -> bf16 head-concat [B,S,D] in ws
// Stage 3: out = H @ W_o^T + b_o -> f32 d_out

typedef __attribute__((ext_vector_type(16))) __bf16 v16bf;
typedef __attribute__((ext_vector_type(8)))  __bf16 v8bf;
typedef __attribute__((ext_vector_type(8)))  float  v8f;

#define D_MODEL 1024
#define NH      16
#define DK      64
#define SEQ     2048
#define NB      2
#define SMPITCH 2052   // 2048 + 4 floats: conflict-free row-major frag gathers

__device__ __forceinline__ __bf16 bf(float x) { return (__bf16)x; }

__device__ __forceinline__ v16bf cat8(v8bf lo, v8bf hi) {
  return __builtin_shufflevector(lo, hi, 0, 1, 2, 3, 4, 5, 6, 7,
                                         8, 9, 10, 11, 12, 13, 14, 15);
}

__device__ __forceinline__ v8f wmma_bf16(v16bf a, v16bf b, v8f c) {
  return __builtin_amdgcn_wmma_f32_16x16x32_bf16(false, a, false, b,
                                                 (short)0, c, false, false);
}

// ---------------------------------------------------------------- stage 1
// 128-thread block = 4 waves; each wave owns one 16x16 tile of X @ W^T.
// MODE 0: Out[bh][s][dk]  (Q, K)      MODE 1: Out[bh][dk][s]  (V transposed)
template <int MODE>
__global__ void __launch_bounds__(128)
proj_kernel(const float* __restrict__ X, const float* __restrict__ W,
            __bf16* __restrict__ Out) {
  const int lane = threadIdx.x & 31;
  const int wave = threadIdx.x >> 5;
  const int g = lane >> 4, l = lane & 15;
  const int row0 = blockIdx.x * 16;                    // token row tile
  const int col0 = (blockIdx.y * 4 + wave) * 16;       // model-dim tile

  const float* arow = X + (size_t)(row0 + l) * D_MODEL;
  const float* brow = W + (size_t)(col0 + l) * D_MODEL;

  v8f acc = {};
  for (int k0 = 0; k0 < D_MODEL; k0 += 32) {
    v16bf a, b;
#pragma unroll
    for (int j = 0; j < 8; ++j) {
      const int ka = k0 + 2 * (j & 3) + 16 * (j >> 2) + 8 * g;  // A 16x32 layout
      a[2 * j]     = bf(arow[ka]);
      a[2 * j + 1] = bf(arow[ka + 1]);
      const int kb = k0 + 2 * (j & 3) + 16 * (j >> 2) + 8 * g;  // B 32x16 layout
      b[2 * j]     = bf(brow[kb]);
      b[2 * j + 1] = bf(brow[kb + 1]);
    }
    acc = wmma_bf16(a, b, acc);
  }
#pragma unroll
  for (int r = 0; r < 8; ++r) {
    const int row = row0 + r + 8 * g;                 // C layout: M = r + 8g
    const int col = col0 + l;                         // N = lane&15
    const int bi = row >> 11, s = row & (SEQ - 1);
    const int h  = col >> 6,  d = col & (DK - 1);
    const size_t bh = (size_t)bi * NH + h;
    if (MODE == 0)
      Out[(bh * SEQ + s) * DK + d] = bf(acc[r]);
    else
      Out[(bh * DK + d) * SEQ + s] = bf(acc[r]);
  }
}

// ---------------------------------------------------------------- stage 2
// Fused attention: one 256-thread block (8 waves) per (bh, 16-query strip).
__global__ void __launch_bounds__(256)
fused_attn_kernel(const __bf16* __restrict__ Q, const __bf16* __restrict__ K,
                  const __bf16* __restrict__ Vt, float* __restrict__ attn,
                  __bf16* __restrict__ Hout) {
  __shared__ float sm[16][SMPITCH];     // score / prob strip (~128.25 KB)
  __shared__ float wavescr[16][8];      // per-wave row stats
  __shared__ float rowstat[16];         // broadcast row max / inv-sum
  __shared__ float partial[4][16][16];  // pass-2 cross-wave partials

  const int t = threadIdx.x;
  const int wave = t >> 5, lane = t & 31;
  const int g = lane >> 4, l = lane & 15;
  const int row0 = blockIdx.x * 16;
  const size_t bh = blockIdx.y;
  const int bi = (int)(bh >> 4), h = (int)(bh & 15);

  const __bf16* q  = Q  + bh * (size_t)SEQ * DK;
  const __bf16* k  = K  + bh * (size_t)SEQ * DK;
  const __bf16* vt = Vt + bh * (size_t)DK * SEQ;

  // ---- pass 1: score strip via WMMA; K=64 in two 32-chunks ----
  const __bf16* qrow = q + (size_t)(row0 + l) * DK;
  const v16bf a0 = cat8(*(const v8bf*)(qrow + 16 * g),
                        *(const v8bf*)(qrow + 8 + 16 * g));
  const v16bf a1 = cat8(*(const v8bf*)(qrow + 32 + 16 * g),
                        *(const v8bf*)(qrow + 40 + 16 * g));

  for (int ct = wave; ct < SEQ / 16; ct += 8) {
    const __bf16* krow = k + (size_t)(ct * 16 + l) * DK;
    // Speculative prefetch (silently dropped if invalid): next tile this wave.
    __builtin_prefetch(krow + 8 * 16 * DK, 0, 1);
    const v16bf b0 = *(const v16bf*)(krow + 16 * g);
    const v16bf b1 = *(const v16bf*)(krow + 32 + 16 * g);
    v8f acc = {};
    acc = wmma_bf16(a0, b0, acc);
    acc = wmma_bf16(a1, b1, acc);
#pragma unroll
    for (int r = 0; r < 8; ++r)
      sm[r + 8 * g][ct * 16 + l] = acc[r] * 0.125f;   // 1/sqrt(64)
  }
  __syncthreads();

  // ---- softmax over the strip (row-wise over 2048), float4 passes ----
  float pm[16];
#pragma unroll
  for (int r = 0; r < 16; ++r) pm[r] = -3.0e38f;
#pragma unroll
  for (int i = 0; i < 2; ++i) {
    const int c = (t + 256 * i) * 4;
#pragma unroll
    for (int r = 0; r < 16; ++r) {
      const float4 v = *(const float4*)&sm[r][c];
      pm[r] = fmaxf(pm[r], fmaxf(fmaxf(v.x, v.y), fmaxf(v.z, v.w)));
    }
  }
#pragma unroll
  for (int r = 0; r < 16; ++r)
#pragma unroll
    for (int off = 16; off > 0; off >>= 1)
      pm[r] = fmaxf(pm[r], __shfl_xor(pm[r], off, 32));
  if (lane == 0)
#pragma unroll
    for (int r = 0; r < 16; ++r) wavescr[r][wave] = pm[r];
  __syncthreads();
  if (t < 16) {
    float m = wavescr[t][0];
#pragma unroll
    for (int w = 1; w < 8; ++w) m = fmaxf(m, wavescr[t][w]);
    rowstat[t] = m;
  }
  __syncthreads();
  float rmax[16];
#pragma unroll
  for (int r = 0; r < 16; ++r) rmax[r] = rowstat[r];
  __syncthreads();

  float ps[16];
#pragma unroll
  for (int r = 0; r < 16; ++r) ps[r] = 0.f;
#pragma unroll
  for (int i = 0; i < 2; ++i) {
    const int c = (t + 256 * i) * 4;
#pragma unroll
    for (int r = 0; r < 16; ++r) {
      float4 v = *(const float4*)&sm[r][c];
      v.x = __expf(v.x - rmax[r]);
      v.y = __expf(v.y - rmax[r]);
      v.z = __expf(v.z - rmax[r]);
      v.w = __expf(v.w - rmax[r]);
      *(float4*)&sm[r][c] = v;
      ps[r] += (v.x + v.y) + (v.z + v.w);
    }
  }
#pragma unroll
  for (int r = 0; r < 16; ++r)
#pragma unroll
    for (int off = 16; off > 0; off >>= 1)
      ps[r] += __shfl_xor(ps[r], off, 32);
  if (lane == 0)
#pragma unroll
    for (int r = 0; r < 16; ++r) wavescr[r][wave] = ps[r];
  __syncthreads();
  if (t < 16) {
    float s = 0.f;
#pragma unroll
    for (int w = 0; w < 8; ++w) s += wavescr[t][w];
    rowstat[t] = 1.0f / s;
  }
  __syncthreads();
  float rinv[16];
#pragma unroll
  for (int r = 0; r < 16; ++r) rinv[r] = rowstat[r];

  // ---- normalize in LDS + single coalesced float4 attn write to HBM ----
  float* arow_g = attn + bh * (size_t)SEQ * SEQ + (size_t)row0 * SEQ;
#pragma unroll
  for (int i = 0; i < 2; ++i) {
    const int c = (t + 256 * i) * 4;
#pragma unroll
    for (int r = 0; r < 16; ++r) {
      float4 v = *(const float4*)&sm[r][c];
      v.x *= rinv[r]; v.y *= rinv[r]; v.z *= rinv[r]; v.w *= rinv[r];
      *(float4*)&sm[r][c] = v;
      *(float4*)(arow_g + (size_t)r * SEQ + c) = v;
    }
  }
  __syncthreads();

  // ---- pass 2: O strip = P @ V_h from LDS; waves split K, pairs reduce ----
  const int ct2 = wave & 3;              // dk output tile (0..3)
  const int kbase = (wave >> 2) * (SEQ / 2);
  const __bf16* vrow = vt + (size_t)(ct2 * 16 + l) * SEQ;
  const float* prow = &sm[l][0];
  v8f acc = {};
  for (int k0 = kbase; k0 < kbase + SEQ / 2; k0 += 32) {
    // A-frag: one 16-float contiguous run -> 4 x ds_load_b128 + pk converts
    const float4 x0 = *(const float4*)(prow + k0 + 16 * g);
    const float4 x1 = *(const float4*)(prow + k0 + 16 * g + 4);
    const float4 x2 = *(const float4*)(prow + k0 + 16 * g + 8);
    const float4 x3 = *(const float4*)(prow + k0 + 16 * g + 12);
    v16bf a;
    a[0]  = bf(x0.x); a[1]  = bf(x0.y); a[2]  = bf(x0.z); a[3]  = bf(x0.w);
    a[4]  = bf(x1.x); a[5]  = bf(x1.y); a[6]  = bf(x1.z); a[7]  = bf(x1.w);
    a[8]  = bf(x2.x); a[9]  = bf(x2.y); a[10] = bf(x2.z); a[11] = bf(x2.w);
    a[12] = bf(x3.x); a[13] = bf(x3.y); a[14] = bf(x3.z); a[15] = bf(x3.w);
    const v16bf b = *(const v16bf*)(vrow + k0 + 16 * g);
    acc = wmma_bf16(a, b, acc);
  }
  if (wave >= 4) {
#pragma unroll
    for (int r = 0; r < 8; ++r) partial[ct2][r + 8 * g][l] = acc[r];
  }
  __syncthreads();
  if (wave < 4) {
#pragma unroll
    for (int r = 0; r < 8; ++r) {
      const float o = acc[r] + partial[ct2][r + 8 * g][l];
      Hout[((size_t)bi * SEQ + row0 + r + 8 * g) * D_MODEL + h * DK +
           ct2 * 16 + l] = bf(o);
    }
  }
}

// ---------------------------------------------------------------- stage 3
// out = Hout @ W_o^T + b_o  -> f32 d_out. 4 waves/block, one tile each.
__global__ void __launch_bounds__(128)
oproj_kernel(const __bf16* __restrict__ H, const float* __restrict__ Wo,
             const float* __restrict__ bo, float* __restrict__ out) {
  const int lane = threadIdx.x & 31;
  const int wave = threadIdx.x >> 5;
  const int g = lane >> 4, l = lane & 15;
  const int row0 = blockIdx.x * 16;
  const int col0 = (blockIdx.y * 4 + wave) * 16;

  const __bf16* arow = H  + (size_t)(row0 + l) * D_MODEL;
  const float*  brow = Wo + (size_t)(col0 + l) * D_MODEL;

  v8f acc = {};
  for (int k0 = 0; k0 < D_MODEL; k0 += 32) {
    const v16bf a = cat8(*(const v8bf*)(arow + k0 + 8 * g),
                         *(const v8bf*)(arow + k0 + 16 + 8 * g));
    v16bf b;
#pragma unroll
    for (int j = 0; j < 8; ++j) {
      const int kb = k0 + 2 * (j & 3) + 16 * (j >> 2) + 8 * g;
      b[2 * j]     = bf(brow[kb]);
      b[2 * j + 1] = bf(brow[kb + 1]);
    }
    acc = wmma_bf16(a, b, acc);
  }
#pragma unroll
  for (int r = 0; r < 8; ++r) {
    const int row = row0 + r + 8 * g;
    const int col = col0 + l;
    out[(size_t)row * D_MODEL + col] = acc[r] + bo[col];
  }
}

// ---------------------------------------------------------------- launcher
extern "C" void kernel_launch(void* const* d_in, const int* in_sizes, int n_in,
                              void* d_out, int out_size, void* d_ws, size_t ws_size,
                              hipStream_t stream) {
  const float* query = (const float*)d_in[0];
  const float* key   = (const float*)d_in[1];
  const float* value = (const float*)d_in[2];
  const float* w_q   = (const float*)d_in[3];
  const float* w_k   = (const float*)d_in[4];
  const float* w_v   = (const float*)d_in[5];
  const float* w_o   = (const float*)d_in[6];
  const float* b_o   = (const float*)d_in[7];

  float* out  = (float*)d_out;                          // [B,S,D]   f32
  float* attn = out + (size_t)NB * SEQ * D_MODEL;       // [B,H,S,S] f32

  const size_t QKV = (size_t)NB * NH * SEQ * DK;        // 4,194,304 elems
  __bf16* ws = (__bf16*)d_ws;
  __bf16* Qw = ws;                                      // [bh][s][dk]
  __bf16* Kw = ws + QKV;                                // [bh][s][dk]
  __bf16* Vw = ws + 2 * QKV;                            // [bh][dk][s]
  __bf16* Hw = ws + 3 * QKV;                            // [B,S,D] bf16

  dim3 gProj(256, 16);                                  // 4096/16 x 1024/(16*4)
  proj_kernel<0><<<gProj, 128, 0, stream>>>(query, w_q, Qw);
  proj_kernel<0><<<gProj, 128, 0, stream>>>(key,   w_k, Kw);
  proj_kernel<1><<<gProj, 128, 0, stream>>>(value, w_v, Vw);

  fused_attn_kernel<<<dim3(SEQ / 16, NB * NH), 256, 0, stream>>>(Qw, Kw, Vw,
                                                                 attn, Hw);
  oproj_kernel<<<gProj, 128, 0, stream>>>(Hw, w_o, b_o, out);
}
